// Alpha_EMRootSpike_GLM_5617817223519
// MI455X (gfx1250) — matrix operand order repaired and checked by reading, not compiled, not run on hardware
//
#include <hip/hip_runtime.h>
#include <math.h>

// ---------------------------------------------------------------------------
// MI455X (gfx1250, wave32) implementation of the GLM neuron reference.
// Heavy ops mapped to v_wmma_f32_16x16x32_f16:
//   * spike routing GEMMs  S_e @ C_syn_e^T, S_i @ C_syn_i^T   (exact in f16)
//   * grouped causal convs as banded-Toeplitz WMMA accumulation with the
//     Toeplitz B-operands held resident in VGPRs across a 34-tile strip
// Intermediates live in L2 (192 MB); the pipeline is bound by streaming the
// 400 MB f32 spike rasters (~22 us floor at 23.3 TB/s).
// ---------------------------------------------------------------------------

typedef __attribute__((ext_vector_type(16))) _Float16 v16h;
typedef __attribute__((ext_vector_type(8)))  _Float16 v8h;
typedef __attribute__((ext_vector_type(8)))  float    v8f;
typedef __attribute__((ext_vector_type(4)))  float    v4f;

union V16H { v16h v; v8h h[2]; };

// ----- problem constants ----------------------------------------------------
#define T_DATA   200000
#define E_NO     400
#define I_NO     100
#define SUB_NO   20
#define T_NO     200
#define N_BASIS  20

#define TP       200192              // 782 * 256  (T rounded up to conv tiles)
#define ROWH     (256 + TP)          // f16 signal row: 256-left-pad + TP
#define NTILES16 (T_DATA / 16)       // 12500 GEMM time tiles
#define NTILES256 782                // conv tiles
#define NKGRP    21                  // 20 subunits + 1 history group
#define KS_E     13                  // ceil(400/32)
#define KS_I     4                   // ceil(100/32)
#define NSTRIPS  23                  // conv tile strips (23 * 34 == 782)
#define TILES_PER 34

// ----- workspace layout (bytes, all 256-aligned); total ~50.7 MB ------------
#define WS_SYNE   0ull
#define WS_SYNI   (WS_SYNE + 20ull * ROWH * 2)
#define WS_ZBUF   (WS_SYNI + 20ull * ROWH * 2)
#define WS_SNS    (WS_ZBUF + (unsigned long long)ROWH * 2)
#define WS_SS     (WS_SNS  + 20ull * TP * 4)
#define WS_HNS    (WS_SS   + 20ull * TP * 4)
#define WS_HS     (WS_HNS  + (unsigned long long)TP * 4)
#define WS_HBUF   (WS_HS   + (unsigned long long)TP * 4)
#define WS_BE     (WS_HBUF + 82ull * 7 * 512 * 2)
#define WS_BI     (WS_BE   + (unsigned long long)KS_E * 2 * 512 * 2)
#define WS_COSB   (WS_BI   + (unsigned long long)KS_I * 2 * 512 * 2)

static __device__ __forceinline__ v8f wmma_f16(v16h a, v16h b, v8f c) {
  return __builtin_amdgcn_wmma_f32_16x16x32_f16(false, a, false, b,
                                                (short)0, c, false, false);
}

// load 8 consecutive f32, convert to 8 f16 (16B-aligned source)
static __device__ __forceinline__ v8h cvt8(const float* p) {
  v4f x0 = *(const v4f*)p;
  v4f x1 = *(const v4f*)(p + 4);
  v8h r;
#pragma unroll
  for (int i = 0; i < 4; ++i) { r[i] = (_Float16)x0[i]; r[4 + i] = (_Float16)x1[i]; }
  return r;
}

static __device__ __forceinline__ v8h cvt8_guard(const float* row, int k, int kreal) {
  if (k + 8 <= kreal) return cvt8(row + k);
  v8h r;
#pragma unroll
  for (int i = 0; i < 8; ++i)
    r[i] = (k + i < kreal) ? (_Float16)row[k + i] : (_Float16)0.0f;
  return r;
}

// Build f16 A-operand (16x32, M x K) from a time-contiguous f16 signal.
// ISA layout: lanes 0-15 (M=lane) hold K {0..7,16..23}; lanes 16-31 (M=lane-16)
// hold K {8..15,24..31}.  All chunk addresses are 16B aligned.
static __device__ __forceinline__ v16h load_conv_A(const _Float16* x, int base, int lane) {
  const int m  = lane & 15;
  const int lo = (lane < 16);
  V16H a;
  a.h[0] = *(const v8h*)(x + base + 16 * m + (lo ? 0 : 8));
  a.h[1] = *(const v8h*)(x + base + 16 * m + (lo ? 16 : 24));
  return a.v;
}

// ===========================================================================
// k_prep: cosine basis, all 82 filter rows (also the out_filters output),
// Toeplitz H operands, B operands for the spike GEMMs, pad zeroing.
// ===========================================================================
__global__ void k_prep(const float* CsynE, const float* CsynI,
                       const float* Wsyn, const float* Tau, const float* Delta,
                       const float* Wsyns, const float* histSw, const float* histNSw,
                       float* cosB, float* outF,
                       _Float16* Hbuf, _Float16* Be, _Float16* Bi,
                       _Float16* synE, _Float16* synI, _Float16* zbuf) {
  const int tid = threadIdx.x;
  const float PI = 3.14159265358979f;
  // raised-cosine basis (N_BASIS x T_NO) in log time
  for (int e = tid; e < N_BASIS * T_NO; e += 256) {
    int i = e / T_NO, x = e % T_NO;
    float raw = 5.0f * __logf((float)x + 1.0f);
    float phi = 0.5f * PI * (float)i;
    float v = 0.5f * __cosf(raw - phi) + 0.5f;
    cosB[e] = (raw >= phi - PI && raw <= phi + PI) ? v : 0.0f;
  }
  __syncthreads();
  // 82 filter rows -> directly into d_out's out_filters region
  for (int e = tid; e < 82 * T_NO; e += 256) {
    int row = e / T_NO, k = e % T_NO;
    float val;
    if (row < 20) {                       // e_kern_ns
      int s = row; float tau = Tau[s * 2]; tau *= tau;
      float te = fmaxf((float)k - Delta[s * 2], 0.0f) / tau;
      float w = Wsyn[s * 2];
      val = te * __expf(-te) * w * w;
    } else if (row < 40) {                // i_kern_ns (negative)
      int s = row - 20; float tau = Tau[s * 2 + 1]; tau *= tau;
      float ti = fmaxf((float)k - Delta[s * 2 + 1], 0.0f) / tau;
      float w = Wsyn[s * 2 + 1];
      val = -ti * __expf(-ti) * w * w;
    } else if (row < 60) {                // e_kern_s = (W^2) @ cos_basis
      int s = row - 40; float acc = 0.0f;
      for (int b = 0; b < N_BASIS; ++b) { float w = Wsyns[(s * N_BASIS + b) * 2]; acc += w * w * cosB[b * T_NO + k]; }
      val = acc;
    } else if (row < 80) {                // i_kern_s (negative)
      int s = row - 60; float acc = 0.0f;
      for (int b = 0; b < N_BASIS; ++b) { float w = Wsyns[(s * N_BASIS + b) * 2 + 1]; acc += w * w * cosB[b * T_NO + k]; }
      val = -acc;
    } else if (row == 80) {               // hist_ns_kern
      float acc = 0.0f;
      for (int b = 0; b < N_BASIS; ++b) acc += histNSw[b] * cosB[b * T_NO + k];
      val = acc;
    } else {                              // hist_s_kern
      float acc = 0.0f;
      for (int b = 0; b < N_BASIS; ++b) acc += histSw[b] * cosB[b * T_NO + k];
      val = acc;
    }
    outF[e] = val;
  }
  __syncthreads();
  // Banded Toeplitz H in exact B-operand lane layout:
  // H_j[r,n] = h[32j + 16 + n - r], masked to [0,200)
  for (int e = tid; e < 82 * 7 * 512; e += 256) {
    int kern = e / (7 * 512), rem = e % (7 * 512);
    int j = rem / 512, lane = (rem % 512) / 16, hh = rem % 16;
    int n = lane & 15, r = ((lane < 16) ? 0 : 8) + hh + ((hh >= 8) ? 8 : 0);
    int k = 32 * j + 16 + n - r;
    int row = (kern < 80) ? ((kern & 3) * 20 + (kern >> 2)) : kern;
    float val = (k >= 0 && k < T_NO) ? outF[row * T_NO + k] : 0.0f;
    Hbuf[e] = (_Float16)val;
  }
  // B operands for spike routing GEMMs: B[k, col] = C_syn[col][k]
  for (int e = tid; e < KS_E * 2 * 512; e += 256) {
    int ks = e / 1024, rem = e % 1024, nt = rem / 512, lane = (rem % 512) / 16, hh = rem % 16;
    int col = nt * 16 + (lane & 15);
    int k = ks * 32 + ((lane < 16) ? 0 : 8) + hh + ((hh >= 8) ? 8 : 0);
    Be[e] = (col < SUB_NO && k < E_NO) ? (_Float16)CsynE[col * E_NO + k] : (_Float16)0.0f;
  }
  for (int e = tid; e < KS_I * 2 * 512; e += 256) {
    int ks = e / 1024, rem = e % 1024, nt = rem / 512, lane = (rem % 512) / 16, hh = rem % 16;
    int col = nt * 16 + (lane & 15);
    int k = ks * 32 + ((lane < 16) ? 0 : 8) + hh + ((hh >= 8) ? 8 : 0);
    Bi[e] = (col < SUB_NO && k < I_NO) ? (_Float16)CsynI[col * I_NO + k] : (_Float16)0.0f;
  }
  // zero pads: whole Z row; left pad + top pad of each signal row
  for (int e = tid; e < ROWH; e += 256) zbuf[e] = (_Float16)0.0f;
  for (int e = tid; e < 20 * 448; e += 256) {
    int s = e / 448, o = e % 448;
    int idx = (o < 256) ? o : (256 + T_DATA + (o - 256));
    synE[s * ROWH + idx] = (_Float16)0.0f;
    synI[s * ROWH + idx] = (_Float16)0.0f;
  }
}

// ===========================================================================
// k_gemm: syn_e = S_e @ C_syn_e^T, syn_i = S_i @ C_syn_i^T via WMMA f16.
// One wave per 16-time tile; M=16 times, N=32 (20 subunits + pad), K padded.
// Outputs stored time-contiguous per subunit (f16) for the conv stage.
// ===========================================================================
__global__ void k_gemm(const float* __restrict__ Se, const float* __restrict__ Si,
                       const float* __restrict__ Z,
                       const _Float16* __restrict__ Be, const _Float16* __restrict__ Bi,
                       _Float16* __restrict__ synE, _Float16* __restrict__ synI,
                       _Float16* __restrict__ zbuf) {
  const int lane = threadIdx.x & 31;
  const int wid  = threadIdx.x >> 5;
  int tile = blockIdx.x * 8 + wid;
  const bool dup = (tile >= NTILES16);
  if (dup) tile = NTILES16 - 1;
  const int t0 = tile * 16;
  const int trow = t0 + (lane & 15);
  const int lo = (lane < 16);
  const float* rowE = Se + (size_t)trow * E_NO;
  const float* rowI = Si + (size_t)trow * I_NO;

  v8f accE0 = {}, accE1 = {}, accI0 = {}, accI1 = {};
#pragma unroll
  for (int ks = 0; ks < KS_E; ++ks) {
    const int k0 = ks * 32;
    __builtin_prefetch(rowE + k0 + 64, 0, 1);
    V16H a;
    a.h[0] = cvt8_guard(rowE, k0 + (lo ? 0 : 8), E_NO);
    a.h[1] = cvt8_guard(rowE, k0 + (lo ? 16 : 24), E_NO);
    v16h b0 = *(const v16h*)(Be + ((size_t)(ks * 2 + 0) * 32 + lane) * 16);
    v16h b1 = *(const v16h*)(Be + ((size_t)(ks * 2 + 1) * 32 + lane) * 16);
    accE0 = wmma_f16(a.v, b0, accE0);
    accE1 = wmma_f16(a.v, b1, accE1);
  }
#pragma unroll
  for (int ks = 0; ks < KS_I; ++ks) {
    const int k0 = ks * 32;
    V16H a;
    a.h[0] = cvt8_guard(rowI, k0 + (lo ? 0 : 8), I_NO);
    a.h[1] = cvt8_guard(rowI, k0 + (lo ? 16 : 24), I_NO);
    v16h b0 = *(const v16h*)(Bi + ((size_t)(ks * 2 + 0) * 32 + lane) * 16);
    v16h b1 = *(const v16h*)(Bi + ((size_t)(ks * 2 + 1) * 32 + lane) * 16);
    accI0 = wmma_f16(a.v, b0, accI0);
    accI1 = wmma_f16(a.v, b1, accI1);
  }

  if (!dup) {
    const int n0 = lane & 15;
    const int mB = lo ? 0 : 8;           // D rows: lanes<16 -> M=v, else M=v+8
    // n-tile 0: subunits 0..15
    {
      _Float16* de = synE + (size_t)n0 * ROWH + 256 + t0 + mB;
      _Float16* di = synI + (size_t)n0 * ROWH + 256 + t0 + mB;
#pragma unroll
      for (int v = 0; v < 8; ++v) { de[v] = (_Float16)accE0[v]; di[v] = (_Float16)accI0[v]; }
    }
    // n-tile 1: subunits 16..19
    if (n0 < SUB_NO - 16) {
      const int s = 16 + n0;
      _Float16* de = synE + (size_t)s * ROWH + 256 + t0 + mB;
      _Float16* di = synI + (size_t)s * ROWH + 256 + t0 + mB;
#pragma unroll
      for (int v = 0; v < 8; ++v) { de[v] = (_Float16)accE1[v]; di[v] = (_Float16)accI1[v]; }
    }
    // stage Z as f16, shifted by +1 so conv reads give the causal t-1 offset
    if (lane < 16) zbuf[257 + t0 + lane] = (_Float16)Z[t0 + lane];
  }
}

// ===========================================================================
// k_conv: causal 200-tap convs as 7 accumulating 16x16x32 WMMAs per signal.
// Each wave owns one (kernel-group, output-kind) pair and a 34-tile strip.
// Its 14 banded-Toeplitz B-operands are preloaded ONCE into VGPRs (112 VGPRs)
// and stay resident across the strip -- removes ~440 MB of redundant
// B-operand L2 traffic vs. the tile-per-wave formulation.
// Groups 0..19 = subunits (e+i signals fused into one accumulator);
// group 20 = Z spike-history. kind: 0 = "ns", 1 = "s".
// ===========================================================================
__global__ void k_conv(const _Float16* __restrict__ synE, const _Float16* __restrict__ synI,
                       const _Float16* __restrict__ zbuf, const _Float16* __restrict__ H,
                       float* __restrict__ SNS, float* __restrict__ SS,
                       float* __restrict__ histNS, float* __restrict__ histS) {
  const int lane = threadIdx.x & 31;
  const int wid  = threadIdx.x >> 5;
  int job = blockIdx.x * 8 + wid;
  const int nJobs = NKGRP * 2 * NSTRIPS;          // 966
  const bool dup = (job >= nJobs);
  if (dup) job = nJobs - 1;
  const int strip = job % NSTRIPS;
  const int rem   = job / NSTRIPS;
  const int kind  = rem & 1;                      // 0 = ns, 1 = s
  const int kg    = rem >> 1;                     // 0..20

  // ---- preload Toeplitz B-operands into registers (resident across strip)
  v16h He[7], Hi[7];
  if (kg < SUB_NO) {
    const _Float16* he = H + (size_t)(kg * 4 + (kind ? 2 : 0)) * 7 * 512;
    const _Float16* hi = H + (size_t)(kg * 4 + (kind ? 3 : 1)) * 7 * 512;
#pragma unroll
    for (int j = 0; j < 7; ++j) {
      He[j] = *(const v16h*)(he + (size_t)j * 512 + lane * 16);
      Hi[j] = *(const v16h*)(hi + (size_t)j * 512 + lane * 16);
    }
  } else {
    const _Float16* hz = H + (size_t)(kind ? 81 : 80) * 7 * 512;
#pragma unroll
    for (int j = 0; j < 7; ++j) {
      He[j] = *(const v16h*)(hz + (size_t)j * 512 + lane * 16);
      Hi[j] = He[j];                              // unused on hist path
    }
  }

  const _Float16* xe = (kg < SUB_NO) ? (synE + (size_t)kg * ROWH) : zbuf;
  const _Float16* xi = synI + (size_t)((kg < SUB_NO) ? kg : 0) * ROWH;
  float* dstBase = (kg < SUB_NO)
      ? (kind ? (SS + (size_t)kg * TP) : (SNS + (size_t)kg * TP))
      : (kind ? histS : histNS);
  const int loff = (lane < 16) ? lane : (112 + lane);

  const int tEnd = (strip + 1) * TILES_PER;
  for (int tile = strip * TILES_PER; tile < tEnd; ++tile) {
    const int t0 = tile * 256;
    __builtin_prefetch(xe + 256 + t0 + 256, 0, 1);  // next tile's A window
    v8f acc = {};
    if (kg < SUB_NO) {
#pragma unroll
      for (int j = 0; j < 7; ++j) {
        const int base = 256 + t0 - 32 * j - 16;
        acc = wmma_f16(load_conv_A(xe, base, lane), He[j], acc);
        acc = wmma_f16(load_conv_A(xi, base, lane), Hi[j], acc);
      }
    } else {
#pragma unroll
      for (int j = 0; j < 7; ++j) {
        const int base = 256 + t0 - 32 * j - 16;
        acc = wmma_f16(load_conv_A(xe, base, lane), He[j], acc);
      }
    }
    if (!dup) {
      // D[m,n] -> time t0 + 16m + n ; per VGPR both lane halves contiguous
      float* d = dstBase + t0;
#pragma unroll
      for (int v = 0; v < 8; ++v) d[16 * v + loff] = acc[v];
    }
  }
}

// ===========================================================================
// k_final: leaf-to-root binary-tree recursion + sigmoid/tanh outputs.
// C_den / weights / thetas cached in LDS; per-thread time step.
// ===========================================================================
__global__ void k_final(const float* __restrict__ SNS, const float* __restrict__ SS,
                        const float* __restrict__ histNS, const float* __restrict__ histS,
                        const int* __restrict__ Cden,
                        const float* __restrict__ WsubNS, const float* __restrict__ WsubS,
                        const float* __restrict__ Vo,
                        const float* __restrict__ ThNS, const float* __restrict__ ThS,
                        float* __restrict__ out) {
  __shared__ float s_ws2[SUB_NO], s_wns2[SUB_NO], s_ths[SUB_NO], s_thns[SUB_NO];
  __shared__ int s_cden[SUB_NO * SUB_NO];
  const int tid = threadIdx.x;
  if (tid < SUB_NO) {
    float a = WsubS[tid], b = WsubNS[tid];
    s_ws2[tid] = a * a; s_wns2[tid] = b * b;
    s_ths[tid] = ThS[tid]; s_thns[tid] = ThNS[tid];
  }
  for (int i = tid; i < SUB_NO * SUB_NO; i += 256) s_cden[i] = Cden[i];
  __syncthreads();

  const int t = blockIdx.x * 256 + tid;
  if (t >= T_DATA) return;

  float ss[SUB_NO], sns[SUB_NO], os[SUB_NO], ons[SUB_NO];
#pragma unroll
  for (int s = 0; s < SUB_NO; ++s) {
    ss[s]  = SS[(size_t)s * TP + t];
    sns[s] = SNS[(size_t)s * TP + t];
    os[s] = 0.0f; ons[s] = 0.0f;
  }
#pragma unroll
  for (int idx = SUB_NO - 1; idx >= 1; --idx) {
    float cs = 0.0f, cns = 0.0f;
#pragma unroll
    for (int c = 0; c < SUB_NO; ++c) {
      if (s_cden[idx * SUB_NO + c]) { cs += os[c] * s_ws2[c]; cns += ons[c] * s_wns2[c]; }
    }
    os[idx]  = tanhf(ss[idx] + cs + s_ths[idx]);
    ons[idx] = tanhf(sns[idx] + cns + s_thns[idx]);
  }
  float cs = 0.0f, cns = 0.0f;
#pragma unroll
  for (int c = 0; c < SUB_NO; ++c) {
    if (s_cden[c]) { cs += os[c] * s_ws2[c]; cns += ons[c] * s_wns2[c]; }
  }
  float s0  = 1.0f / (1.0f + __expf(-(histS[t] + ss[0] + cs + s_ths[0])));
  float ns0 = tanhf(histNS[t] + sns[0] + cns + s_thns[0]);
  out[t]          = ns0 * s_wns2[0] + Vo[0];   // final_V
  out[T_DATA + t] = s0;                        // final_Z
}

// ===========================================================================
extern "C" void kernel_launch(void* const* d_in, const int* in_sizes, int n_in,
                              void* d_out, int out_size, void* d_ws, size_t ws_size,
                              hipStream_t stream) {
  const float* Se     = (const float*)d_in[0];
  const float* Si     = (const float*)d_in[1];
  const float* Z      = (const float*)d_in[2];
  const int*   Cden   = (const int*)d_in[3];
  const float* CsynE  = (const float*)d_in[4];
  const float* CsynI  = (const float*)d_in[5];
  const float* Wsyn   = (const float*)d_in[6];
  const float* Tau    = (const float*)d_in[7];
  const float* Delta  = (const float*)d_in[8];
  const float* Wsyns  = (const float*)d_in[9];
  const float* WsubNS = (const float*)d_in[10];
  const float* WsubS  = (const float*)d_in[11];
  const float* Vo     = (const float*)d_in[12];
  const float* ThNS   = (const float*)d_in[13];
  const float* ThS    = (const float*)d_in[14];
  const float* histSw = (const float*)d_in[15];
  const float* histNSw= (const float*)d_in[16];

  float* out = (float*)d_out;
  char* ws = (char*)d_ws;
  _Float16* synE = (_Float16*)(ws + WS_SYNE);
  _Float16* synI = (_Float16*)(ws + WS_SYNI);
  _Float16* zbuf = (_Float16*)(ws + WS_ZBUF);
  float*    SNS  = (float*)(ws + WS_SNS);
  float*    SSb  = (float*)(ws + WS_SS);
  float*    hNS  = (float*)(ws + WS_HNS);
  float*    hS   = (float*)(ws + WS_HS);
  _Float16* Hbuf = (_Float16*)(ws + WS_HBUF);
  _Float16* Be   = (_Float16*)(ws + WS_BE);
  _Float16* Bi   = (_Float16*)(ws + WS_BI);
  float*    cosB = (float*)(ws + WS_COSB);
  float*    outF = out + 2 * T_DATA;          // out_filters region of d_out

  k_prep<<<1, 256, 0, stream>>>(CsynE, CsynI, Wsyn, Tau, Delta, Wsyns,
                                histSw, histNSw, cosB, outF,
                                Hbuf, Be, Bi, synE, synI, zbuf);

  const int gemmBlocks = (NTILES16 + 7) / 8;          // 8 waves / block
  k_gemm<<<gemmBlocks, 256, 0, stream>>>(Se, Si, Z, Be, Bi, synE, synI, zbuf);

  const int convBlocks = (NKGRP * 2 * NSTRIPS + 7) / 8;
  k_conv<<<convBlocks, 256, 0, stream>>>(synE, synI, zbuf, Hbuf, SNS, SSb, hNS, hS);

  k_final<<<NTILES256, 256, 0, stream>>>(SNS, SSb, hNS, hS, Cden,
                                         WsubNS, WsubS, Vo, ThNS, ThS, out);
}
